// TrustAwareSelfAttention_81793357185433
// MI455X (gfx1250) — matrix operand order repaired and checked by reading, not compiled, not run on hardware
//
#include <hip/hip_runtime.h>
#include <hip/hip_bf16.h>

typedef __attribute__((ext_vector_type(16))) _Float16 v16h;
typedef __attribute__((ext_vector_type(8)))  _Float16 v8h;
typedef __attribute__((ext_vector_type(8)))  float    v8f;
typedef int i32x4g __attribute__((vector_size(16)));   // matches builtin param type

#define B_   4
#define S_   1024
#define D_   1024
#define H_   16
#define HD_  64
#define BS_  4096          // B_*S_
#define LST  72            // LDS tile K-stride in halves (64 + 8 pad, keeps 16B align)

// ---------------- async global->LDS copy (CDNA5), guarded -------------------

#if defined(__AMDGCN__) && __has_builtin(__builtin_amdgcn_global_load_async_to_lds_b128)
#define ASYNC_MODE 1
#elif defined(__AMDGCN__) && __has_builtin(__builtin_amdgcn_load_to_lds)
#define ASYNC_MODE 2
#else
#define ASYNC_MODE 0
#endif

__device__ __forceinline__ void tile_cp_b128(_Float16* ldst, const _Float16* gsrc) {
#if ASYNC_MODE == 1
  __attribute__((address_space(1))) void* gv =
      (__attribute__((address_space(1))) void*)const_cast<_Float16*>(gsrc);
  __attribute__((address_space(3))) void* lv =
      (__attribute__((address_space(3))) void*)ldst;
  __builtin_amdgcn_global_load_async_to_lds_b128(
      (__attribute__((address_space(1))) i32x4g*)gv,
      (__attribute__((address_space(3))) i32x4g*)lv, 0, 0);
#elif ASYNC_MODE == 2
  __builtin_amdgcn_load_to_lds(
      (__attribute__((address_space(1))) void*)const_cast<_Float16*>(gsrc),
      (__attribute__((address_space(3))) void*)ldst, 16, 0, 0);
#else
  *(v8h*)ldst = *(const v8h*)gsrc;
#endif
}

// Wait until at most N async global->LDS ops remain outstanding (per wave,
// in-order completion => N = loads of the *next* tile leaves current tile done).
template <int N>
__device__ __forceinline__ void tile_cp_wait() {
#if ASYNC_MODE == 1
#if __has_builtin(__builtin_amdgcn_s_wait_asynccnt)
  __builtin_amdgcn_s_wait_asynccnt(N);
#else
  asm volatile("s_wait_asynccnt %0" :: "i"(N) : "memory");
#endif
#elif ASYNC_MODE == 2
  asm volatile("s_wait_asynccnt %0" :: "i"(N) : "memory");
#endif
}

// ---------------- WMMA helpers (CDNA5 16x16x32 f16 -> f32) ------------------

__device__ __forceinline__ v8f wmma_f16(v16h a, v16h b, v8f c) {
  return __builtin_amdgcn_wmma_f32_16x16x32_f16(false, a, false, b, (short)0, c,
                                                false, false);
}

// A fragment: 16x32, lane holds row M=lane%16; halves 0..7 -> K=(lane/16)*8+i,
// halves 8..15 -> K=16+(lane/16)*8+i.
__device__ __forceinline__ v16h fragA(const _Float16* lds, int mrow, int kstep, int lane) {
  const _Float16* p = lds + mrow * LST + kstep + ((lane >> 4) << 3);
  v8h lo = *(const v8h*)p;
  v8h hi = *(const v8h*)(p + 16);
  v16h r;
#pragma unroll
  for (int i = 0; i < 8; ++i) { r[i] = lo[i]; r[i + 8] = hi[i]; }
  return r;
}

// B fragment: 32x16, lane holds col N=lane%16; halves 0..7 -> K=(lane/16)*8+i,
// halves 8..15 -> K=16+(lane/16)*8+i.  LDS B tile is N-major with K-stride LST.
__device__ __forceinline__ v16h fragB(const _Float16* lds, int ncol, int kstep, int lane) {
  const _Float16* p = lds + ncol * LST + kstep + ((lane >> 4) << 3);
  v8h lo = *(const v8h*)p;
  v8h hi = *(const v8h*)(p + 16);
  v16h r;
#pragma unroll
  for (int i = 0; i < 8; ++i) { r[i] = lo[i]; r[i + 8] = hi[i]; }
  return r;
}

__device__ __forceinline__ float sigmoidf_(float x) {
  return 1.0f / (1.0f + __expf(-x));
}

// ---------------- small utility kernels -------------------------------------

__global__ void k_f32_to_f16(const float* __restrict__ src, _Float16* __restrict__ dst, int n) {
  int i = blockIdx.x * blockDim.x + threadIdx.x;
  int stride = gridDim.x * blockDim.x;
  for (; i < n; i += stride) dst[i] = (_Float16)src[i];
}

// Fused convert + transpose: W f32 [K=1024][N=1024] -> Wt f16 [N][K].
__global__ __launch_bounds__(256) void k_cvt_tr_w(const float* __restrict__ W,
                                                  _Float16* __restrict__ Wt) {
  __shared__ _Float16 T[64 * LST];
  const int tid = threadIdx.x;
  const int bk0 = blockIdx.x * 64, bn0 = blockIdx.y * 64;
#pragma unroll
  for (int i = 0; i < 4; ++i) {            // 64x64 f32 tile in, f16 to LDS
    int seg = tid + i * 256;
    int row = seg >> 4, cb = (seg & 15) << 2;
    float4 v = *(const float4*)&W[(size_t)(bk0 + row) * D_ + bn0 + cb];
    T[row * LST + cb + 0] = (_Float16)v.x;
    T[row * LST + cb + 1] = (_Float16)v.y;
    T[row * LST + cb + 2] = (_Float16)v.z;
    T[row * LST + cb + 3] = (_Float16)v.w;
  }
  __syncthreads();
#pragma unroll
  for (int i = 0; i < 2; ++i) {            // transposed rows out, 16B stores
    int seg = tid + i * 256;
    int n = seg >> 3, kb = (seg & 7) << 3;
    v8h o;
#pragma unroll
    for (int j = 0; j < 8; ++j) o[j] = T[(kb + j) * LST + n];
    *(v8h*)&Wt[(size_t)(bn0 + n) * D_ + bk0 + kb] = o;
  }
}

// trust-affinity scalars: ww = W.W, wb = W.b, bb = b.b  (hd = 64)
__global__ void k_tp_stats(const float* __restrict__ Wtp, const float* __restrict__ btp,
                           float* __restrict__ stats) {
  int lane = threadIdx.x;  // 32 threads
  float ww = 0.f, wb = 0.f, bb = 0.f;
  for (int i = lane; i < HD_; i += 32) {
    float w = Wtp[i], b = btp[i];
    ww += w * w; wb += w * b; bb += b * b;
  }
  for (int o = 16; o; o >>= 1) {
    ww += __shfl_xor(ww, o, 32);
    wb += __shfl_xor(wb, o, 32);
    bb += __shfl_xor(bb, o, 32);
  }
  if (lane == 0) { stats[0] = ww; stats[1] = wb; stats[2] = bb; }
}

// ---------------- QKV projection: x(4096x1024) @ W + b ----------------------
// Double-buffered async pipeline: tile k+1 streams into LDS during WMMA of
// tile k (ASYNCcnt in-order completion => wait<=8 retires current tile).
__global__ __launch_bounds__(256) void k_qkv(
    const _Float16* __restrict__ xh,
    const _Float16* __restrict__ Wqt, const _Float16* __restrict__ Wkt,
    const _Float16* __restrict__ Wvt,
    const float* __restrict__ bq, const float* __restrict__ bk, const float* __restrict__ bv,
    _Float16* __restrict__ Qh, _Float16* __restrict__ Kh, _Float16* __restrict__ Vt) {
  __shared__ _Float16 Alds[2][128 * LST];
  __shared__ _Float16 Blds[2][128 * LST];
  const int tid = threadIdx.x, lane = tid & 31, wave = tid >> 5;
  const int bm = blockIdx.y * 128, bn = blockIdx.x * 128;
  const int z = blockIdx.z;
  const _Float16* Wt = (z == 0) ? Wqt : (z == 1) ? Wkt : Wvt;
  const float* bias = (z == 0) ? bq : (z == 1) ? bk : bv;
  const int waveM = wave >> 1, waveN = wave & 1;  // 4x2 waves, wave tile 32x64

  v8f zero = {0.f,0.f,0.f,0.f,0.f,0.f,0.f,0.f};
  v8f acc[2][4];
#pragma unroll
  for (int r = 0; r < 2; ++r)
#pragma unroll
    for (int c = 0; c < 4; ++c) acc[r][c] = zero;

  const int NT = D_ / 64;
#pragma unroll
  for (int i = 0; i < 4; ++i) {       // prologue: tile 0 -> buffer 0
    int seg = tid + i * 256;
    int row = seg >> 3, cb = (seg & 7) << 3;
    tile_cp_b128(&Alds[0][row * LST + cb], &xh[(size_t)(bm + row) * D_ + cb]);
    tile_cp_b128(&Blds[0][row * LST + cb], &Wt[(size_t)(bn + row) * D_ + cb]);
  }
  for (int ti = 0; ti < NT; ++ti) {
    const int cur = ti & 1, nxt = cur ^ 1;
    if (ti + 1 < NT) {
      const int kt = (ti + 1) * 64;
#pragma unroll
      for (int i = 0; i < 4; ++i) {   // prefetch tile ti+1 while computing ti
        int seg = tid + i * 256;
        int row = seg >> 3, cb = (seg & 7) << 3;
        tile_cp_b128(&Alds[nxt][row * LST + cb], &xh[(size_t)(bm + row) * D_ + kt + cb]);
        tile_cp_b128(&Blds[nxt][row * LST + cb], &Wt[(size_t)(bn + row) * D_ + kt + cb]);
      }
      tile_cp_wait<8>();
    } else {
      tile_cp_wait<0>();
    }
    __syncthreads();
#pragma unroll
    for (int ks = 0; ks < 64; ks += 32) {
      v16h a[2], b2[4];
#pragma unroll
      for (int r = 0; r < 2; ++r)
        a[r] = fragA(Alds[cur], waveM * 32 + r * 16 + (lane & 15), ks, lane);
#pragma unroll
      for (int c = 0; c < 4; ++c)
        b2[c] = fragB(Blds[cur], waveN * 64 + c * 16 + (lane & 15), ks, lane);
#pragma unroll
      for (int r = 0; r < 2; ++r)
#pragma unroll
        for (int c = 0; c < 4; ++c) acc[r][c] = wmma_f16(a[r], b2[c], acc[r][c]);
    }
    __syncthreads();
  }
#pragma unroll
  for (int r = 0; r < 2; ++r)
#pragma unroll
    for (int c = 0; c < 4; ++c) {
      int n = bn + waveN * 64 + c * 16 + (lane & 15);
      int h = n >> 6, hd = n & 63;
      float bv_ = bias[n];
#pragma unroll
      for (int e = 0; e < 8; ++e) {
        int m = bm + waveM * 32 + r * 16 + ((lane >> 4) << 3) + e;
        int bat = m >> 10, srow = m & 1023;
        _Float16 val = (_Float16)(acc[r][c][e] + bv_);
        if (z == 2) {
          Vt[((size_t)(bat * H_ + h) * HD_ + hd) * S_ + srow] = val;   // [bh][hd][s]
        } else {
          _Float16* dst = (z == 0) ? Qh : Kh;
          dst[((size_t)(bat * H_ + h) * S_ + srow) * HD_ + hd] = val;  // [bh][s][hd]
        }
      }
    }
}

// ---------------- scores: S = Q K^T / 8 + 0.1*trust_aff, query mask ---------
__global__ __launch_bounds__(256) void k_scores(
    const _Float16* __restrict__ Qh, const _Float16* __restrict__ Kh,
    const float* __restrict__ trust, const int* __restrict__ mask,
    const float* __restrict__ stats, float* __restrict__ attn) {
  __shared__ _Float16 Alds[128 * LST];
  __shared__ _Float16 Blds[128 * LST];
  const int tid = threadIdx.x, lane = tid & 31, wave = tid >> 5;
  const int bh = blockIdx.z, b = bh >> 4;
  const int bm = blockIdx.y * 128, bn = blockIdx.x * 128;
  const int waveM = wave >> 1, waveN = wave & 1;

#pragma unroll
  for (int i = 0; i < 4; ++i) {   // Q tile 128x64 (full K) + K tile (N-major)
    int seg = tid + i * 256;
    int row = seg >> 3, cb = (seg & 7) << 3;
    tile_cp_b128(&Alds[row * LST + cb], &Qh[((size_t)bh * S_ + bm + row) * HD_ + cb]);
    tile_cp_b128(&Blds[row * LST + cb], &Kh[((size_t)bh * S_ + bn + row) * HD_ + cb]);
  }
  tile_cp_wait<0>();
  __syncthreads();

  v8f zero = {0.f,0.f,0.f,0.f,0.f,0.f,0.f,0.f};
  v8f acc[2][4];
#pragma unroll
  for (int r = 0; r < 2; ++r)
#pragma unroll
    for (int c = 0; c < 4; ++c) acc[r][c] = zero;
#pragma unroll
  for (int ks = 0; ks < 64; ks += 32) {
    v16h a[2], b2[4];
#pragma unroll
    for (int r = 0; r < 2; ++r)
      a[r] = fragA(Alds, waveM * 32 + r * 16 + (lane & 15), ks, lane);
#pragma unroll
    for (int c = 0; c < 4; ++c)
      b2[c] = fragB(Blds, waveN * 64 + c * 16 + (lane & 15), ks, lane);
#pragma unroll
    for (int r = 0; r < 2; ++r)
#pragma unroll
      for (int c = 0; c < 4; ++c) acc[r][c] = wmma_f16(a[r], b2[c], acc[r][c]);
  }

  const float ww = stats[0], wb = stats[1], bb = stats[2];
#pragma unroll
  for (int r = 0; r < 2; ++r)
#pragma unroll
    for (int c = 0; c < 4; ++c) {
      int n = bn + waveN * 64 + c * 16 + (lane & 15);
      float tk = trust[b * S_ + n];
#pragma unroll
      for (int e = 0; e < 8; ++e) {
        int m = bm + waveM * 32 + r * 16 + ((lane >> 4) << 3) + e;
        float tq = trust[b * S_ + m];
        float sc = acc[r][c][e] * 0.125f;  // 1/sqrt(64)
        sc += 0.1f * sigmoidf_(tq * tk * ww + (tq + tk) * wb + bb);
        if (mask[b * S_ + m] == 0) sc = -1e9f;
        attn[((size_t)bh * S_ + m) * S_ + n] = sc;
      }
    }
}

// ---------------- per-row softmax stats: max and 1/sum(exp) -----------------
__global__ __launch_bounds__(256) void k_rowstats(
    const float* __restrict__ attn, float* __restrict__ rowmax, float* __restrict__ rowinv) {
  const int lane = threadIdx.x & 31;
  const size_t row = (size_t)blockIdx.x * 8 + (threadIdx.x >> 5);
  const float4* rp = (const float4*)(attn + row * S_);
  float4 v[8];
  float m = -3.4e38f;
#pragma unroll
  for (int i = 0; i < 8; ++i) {
    v[i] = rp[lane + i * 32];
    m = fmaxf(m, fmaxf(fmaxf(v[i].x, v[i].y), fmaxf(v[i].z, v[i].w)));
  }
  for (int o = 16; o; o >>= 1) m = fmaxf(m, __shfl_xor(m, o, 32));
  float s = 0.f;
#pragma unroll
  for (int i = 0; i < 8; ++i) {
    s += __expf(v[i].x - m) + __expf(v[i].y - m) + __expf(v[i].z - m) + __expf(v[i].w - m);
  }
  for (int o = 16; o; o >>= 1) s += __shfl_xor(s, o, 32);
  if (lane == 0) { rowmax[row] = m; rowinv[row] = 1.0f / s; }
}

// ---------------- fused normalize + attended = softmax(S) @ V ---------------
// V tiles double-buffered async; issue order puts the V fetch of tile k+1
// under the expf/normalize VALU work of tile k.
__global__ __launch_bounds__(256) void k_av(
    const _Float16* __restrict__ Vt, const float* __restrict__ rowmax,
    const float* __restrict__ rowinv, float* __restrict__ attn,
    _Float16* __restrict__ attOut) {
  __shared__ _Float16 Alds[128 * LST];
  __shared__ _Float16 Blds[2][64 * LST];
  const int tid = threadIdx.x, lane = tid & 31, wave = tid >> 5;
  const int bh = blockIdx.y, b = bh >> 4, h = bh & 15;
  const int bm = blockIdx.x * 128;
  const int waveM = wave >> 1, waveN = wave & 1;  // wave tile 32x32, N total 64

  v8f zero = {0.f,0.f,0.f,0.f,0.f,0.f,0.f,0.f};
  v8f acc[2][2];
#pragma unroll
  for (int r = 0; r < 2; ++r)
#pragma unroll
    for (int c = 0; c < 2; ++c) acc[r][c] = zero;

  const int NT = S_ / 64;
#pragma unroll
  for (int i = 0; i < 2; ++i) {       // prologue: V tile 0 -> buffer 0
    int seg = tid + i * 256;
    int row = seg >> 3, cb = (seg & 7) << 3;
    tile_cp_b128(&Blds[0][row * LST + cb], &Vt[((size_t)bh * HD_ + row) * S_ + cb]);
  }
  for (int ti = 0; ti < NT; ++ti) {
    const int cur = ti & 1, nxt = cur ^ 1;
    const int kt = ti * 64;
    if (ti + 1 < NT) {
#pragma unroll
      for (int i = 0; i < 2; ++i) {   // prefetch V tile ti+1
        int seg = tid + i * 256;
        int row = seg >> 3, cb = (seg & 7) << 3;
        tile_cp_b128(&Blds[nxt][row * LST + cb],
                     &Vt[((size_t)bh * HD_ + row) * S_ + kt + 64 + cb]);
      }
    }
#pragma unroll
    for (int i = 0; i < 8; ++i) {     // scores tile 128x64: normalize+write+stash
      int seg = tid + i * 256;
      int row = seg >> 4, cb = (seg & 15) << 2;
      size_t grow = (size_t)bh * S_ + bm + row;
      float4* gp = (float4*)&attn[grow * S_ + kt + cb];
      float4 s4 = *gp;
      float rm = rowmax[grow], ri = rowinv[grow];
      float p0 = __expf(s4.x - rm) * ri;
      float p1 = __expf(s4.y - rm) * ri;
      float p2 = __expf(s4.z - rm) * ri;
      float p3 = __expf(s4.w - rm) * ri;
      *gp = make_float4(p0, p1, p2, p3);
      Alds[row * LST + cb + 0] = (_Float16)p0;
      Alds[row * LST + cb + 1] = (_Float16)p1;
      Alds[row * LST + cb + 2] = (_Float16)p2;
      Alds[row * LST + cb + 3] = (_Float16)p3;
    }
    if (ti + 1 < NT) tile_cp_wait<2>(); else tile_cp_wait<0>();
    __syncthreads();
#pragma unroll
    for (int ks = 0; ks < 64; ks += 32) {
      v16h a[2], b2[2];
#pragma unroll
      for (int r = 0; r < 2; ++r)
        a[r] = fragA(Alds, waveM * 32 + r * 16 + (lane & 15), ks, lane);
#pragma unroll
      for (int c = 0; c < 2; ++c)
        b2[c] = fragB(Blds[cur], waveN * 32 + c * 16 + (lane & 15), ks, lane);
#pragma unroll
      for (int r = 0; r < 2; ++r)
#pragma unroll
        for (int c = 0; c < 2; ++c) acc[r][c] = wmma_f16(a[r], b2[c], acc[r][c]);
    }
    __syncthreads();
  }
#pragma unroll
  for (int r = 0; r < 2; ++r)
#pragma unroll
    for (int c = 0; c < 2; ++c) {
      int n = waveN * 32 + c * 16 + (lane & 15);
#pragma unroll
      for (int e = 0; e < 8; ++e) {
        int m = bm + waveM * 32 + r * 16 + ((lane >> 4) << 3) + e;
        attOut[((size_t)(b * S_ + m)) * D_ + h * HD_ + n] = (_Float16)acc[r][c][e];
      }
    }
}

// ---------------- gate: sigmoid(attended @ Wg[:1024] + t*Wg[1024] + bg) -----
__global__ __launch_bounds__(256) void k_gate(
    const _Float16* __restrict__ Ah, const _Float16* __restrict__ Wgt,
    const float* __restrict__ WgLast, const float* __restrict__ bg,
    const float* __restrict__ trust, _Float16* __restrict__ gated) {
  __shared__ _Float16 Alds[2][128 * LST];
  __shared__ _Float16 Blds[2][128 * LST];
  const int tid = threadIdx.x, lane = tid & 31, wave = tid >> 5;
  const int bm = blockIdx.y * 128, bn = blockIdx.x * 128;
  const int waveM = wave >> 1, waveN = wave & 1;

  v8f zero = {0.f,0.f,0.f,0.f,0.f,0.f,0.f,0.f};
  v8f acc[2][4];
#pragma unroll
  for (int r = 0; r < 2; ++r)
#pragma unroll
    for (int c = 0; c < 4; ++c) acc[r][c] = zero;

  const int NT = D_ / 64;
#pragma unroll
  for (int i = 0; i < 4; ++i) {
    int seg = tid + i * 256;
    int row = seg >> 3, cb = (seg & 7) << 3;
    tile_cp_b128(&Alds[0][row * LST + cb], &Ah[(size_t)(bm + row) * D_ + cb]);
    tile_cp_b128(&Blds[0][row * LST + cb], &Wgt[(size_t)(bn + row) * D_ + cb]);
  }
  for (int ti = 0; ti < NT; ++ti) {
    const int cur = ti & 1, nxt = cur ^ 1;
    if (ti + 1 < NT) {
      const int kt = (ti + 1) * 64;
#pragma unroll
      for (int i = 0; i < 4; ++i) {
        int seg = tid + i * 256;
        int row = seg >> 3, cb = (seg & 7) << 3;
        tile_cp_b128(&Alds[nxt][row * LST + cb], &Ah[(size_t)(bm + row) * D_ + kt + cb]);
        tile_cp_b128(&Blds[nxt][row * LST + cb], &Wgt[(size_t)(bn + row) * D_ + kt + cb]);
      }
      tile_cp_wait<8>();
    } else {
      tile_cp_wait<0>();
    }
    __syncthreads();
#pragma unroll
    for (int ks = 0; ks < 64; ks += 32) {
      v16h a[2], b2[4];
#pragma unroll
      for (int r = 0; r < 2; ++r)
        a[r] = fragA(Alds[cur], waveM * 32 + r * 16 + (lane & 15), ks, lane);
#pragma unroll
      for (int c = 0; c < 4; ++c)
        b2[c] = fragB(Blds[cur], waveN * 64 + c * 16 + (lane & 15), ks, lane);
#pragma unroll
      for (int r = 0; r < 2; ++r)
#pragma unroll
        for (int c = 0; c < 4; ++c) acc[r][c] = wmma_f16(a[r], b2[c], acc[r][c]);
    }
    __syncthreads();
  }
#pragma unroll
  for (int r = 0; r < 2; ++r)
#pragma unroll
    for (int c = 0; c < 4; ++c) {
      int n = bn + waveN * 64 + c * 16 + (lane & 15);
      float wl = WgLast[n], bgn = bg[n];
#pragma unroll
      for (int e = 0; e < 8; ++e) {
        int m = bm + waveM * 32 + r * 16 + ((lane >> 4) << 3) + e;
        int bat = m >> 10, srow = m & 1023;
        float gl = acc[r][c][e] + trust[bat * S_ + srow] * wl + bgn;
        float g = sigmoidf_(gl);
        float att = (float)Ah[(size_t)m * D_ + n];
        gated[(size_t)m * D_ + n] = (_Float16)(att * g);
      }
    }
}

// ---------------- output projection: gated @ Wo + bo -> f32 out -------------
__global__ __launch_bounds__(256) void k_out(
    const _Float16* __restrict__ Gh, const _Float16* __restrict__ Wot,
    const float* __restrict__ bo, float* __restrict__ out) {
  __shared__ _Float16 Alds[2][128 * LST];
  __shared__ _Float16 Blds[2][128 * LST];
  const int tid = threadIdx.x, lane = tid & 31, wave = tid >> 5;
  const int bm = blockIdx.y * 128, bn = blockIdx.x * 128;
  const int waveM = wave >> 1, waveN = wave & 1;

  v8f zero = {0.f,0.f,0.f,0.f,0.f,0.f,0.f,0.f};
  v8f acc[2][4];
#pragma unroll
  for (int r = 0; r < 2; ++r)
#pragma unroll
    for (int c = 0; c < 4; ++c) acc[r][c] = zero;

  const int NT = D_ / 64;
#pragma unroll
  for (int i = 0; i < 4; ++i) {
    int seg = tid + i * 256;
    int row = seg >> 3, cb = (seg & 7) << 3;
    tile_cp_b128(&Alds[0][row * LST + cb], &Gh[(size_t)(bm + row) * D_ + cb]);
    tile_cp_b128(&Blds[0][row * LST + cb], &Wot[(size_t)(bn + row) * D_ + cb]);
  }
  for (int ti = 0; ti < NT; ++ti) {
    const int cur = ti & 1, nxt = cur ^ 1;
    if (ti + 1 < NT) {
      const int kt = (ti + 1) * 64;
#pragma unroll
      for (int i = 0; i < 4; ++i) {
        int seg = tid + i * 256;
        int row = seg >> 3, cb = (seg & 7) << 3;
        tile_cp_b128(&Alds[nxt][row * LST + cb], &Gh[(size_t)(bm + row) * D_ + kt + cb]);
        tile_cp_b128(&Blds[nxt][row * LST + cb], &Wot[(size_t)(bn + row) * D_ + kt + cb]);
      }
      tile_cp_wait<8>();
    } else {
      tile_cp_wait<0>();
    }
    __syncthreads();
#pragma unroll
    for (int ks = 0; ks < 64; ks += 32) {
      v16h a[2], b2[4];
#pragma unroll
      for (int r = 0; r < 2; ++r)
        a[r] = fragA(Alds[cur], waveM * 32 + r * 16 + (lane & 15), ks, lane);
#pragma unroll
      for (int c = 0; c < 4; ++c)
        b2[c] = fragB(Blds[cur], waveN * 64 + c * 16 + (lane & 15), ks, lane);
#pragma unroll
      for (int r = 0; r < 2; ++r)
#pragma unroll
        for (int c = 0; c < 4; ++c) acc[r][c] = wmma_f16(a[r], b2[c], acc[r][c]);
    }
    __syncthreads();
  }
#pragma unroll
  for (int r = 0; r < 2; ++r)
#pragma unroll
    for (int c = 0; c < 4; ++c) {
      int n = bn + waveN * 64 + c * 16 + (lane & 15);
      float bon = bo[n];
#pragma unroll
      for (int e = 0; e < 8; ++e) {
        int m = bm + waveM * 32 + r * 16 + ((lane >> 4) << 3) + e;
        out[(size_t)m * D_ + n] = acc[r][c][e] + bon;
      }
    }
}

// ---------------- launch ----------------------------------------------------

extern "C" void kernel_launch(void* const* d_in, const int* in_sizes, int n_in,
                              void* d_out, int out_size, void* d_ws, size_t ws_size,
                              hipStream_t stream) {
  (void)in_sizes; (void)n_in; (void)out_size; (void)ws_size;
  const float* x     = (const float*)d_in[0];
  const float* trust = (const float*)d_in[1];
  const int*   mask  = (const int*)d_in[2];
  const float* Wq = (const float*)d_in[3];  const float* bq = (const float*)d_in[4];
  const float* Wk = (const float*)d_in[5];  const float* bk = (const float*)d_in[6];
  const float* Wv = (const float*)d_in[7];  const float* bv = (const float*)d_in[8];
  const float* Wtp = (const float*)d_in[9]; const float* btp = (const float*)d_in[10];
  const float* Wg = (const float*)d_in[11]; const float* bg = (const float*)d_in[12];
  const float* Wo = (const float*)d_in[13]; const float* bo = (const float*)d_in[14];

  float* out  = (float*)d_out;                 // [B,S,D] f32
  float* attn = out + (size_t)BS_ * D_;        // [B,H,S,S] f32

  char* w = (char*)d_ws;
  const size_t MB = 1u << 20;
  _Float16* xh   = (_Float16*)(w + 0 * MB);    // 8 MB
  _Float16* Wqt  = (_Float16*)(w + 8 * MB);    // 2 MB each, transposed [N][K]
  _Float16* Wkt  = (_Float16*)(w + 10 * MB);
  _Float16* Wvt  = (_Float16*)(w + 12 * MB);
  _Float16* Wot  = (_Float16*)(w + 14 * MB);
  _Float16* Wgt  = (_Float16*)(w + 16 * MB);
  _Float16* Qh   = (_Float16*)(w + 18 * MB);   // [B,H,S,HD], 8 MB
  _Float16* Kh   = (_Float16*)(w + 26 * MB);   // [B,H,S,HD], 8 MB
  _Float16* Vt   = (_Float16*)(w + 34 * MB);   // [B,H,HD,S], 8 MB
  _Float16* attOut = (_Float16*)(w + 42 * MB); // attended f16 [B,S,D], 8 MB
  _Float16* gated  = (_Float16*)(w + 50 * MB); // 8 MB
  float* rowmax = (float*)(w + 58 * MB);       // 256 KB
  float* rowinv = (float*)(w + 58 * MB + 512 * 1024);
  float* stats  = (float*)(w + 59 * MB);

  dim3 blk(256);
  k_f32_to_f16<<<dim3(2048), blk, 0, stream>>>(x, xh, BS_ * D_);
  dim3 trg(16, 16);
  k_cvt_tr_w<<<trg, blk, 0, stream>>>(Wq, Wqt);
  k_cvt_tr_w<<<trg, blk, 0, stream>>>(Wk, Wkt);
  k_cvt_tr_w<<<trg, blk, 0, stream>>>(Wv, Wvt);
  k_cvt_tr_w<<<trg, blk, 0, stream>>>(Wo, Wot);
  k_cvt_tr_w<<<trg, blk, 0, stream>>>(Wg, Wgt);   // first 1024 rows of Wg
  k_tp_stats<<<1, 32, 0, stream>>>(Wtp, btp, stats);

  k_qkv<<<dim3(8, 32, 3), blk, 0, stream>>>(xh, Wqt, Wkt, Wvt, bq, bk, bv, Qh, Kh, Vt);
  k_scores<<<dim3(8, 8, B_ * H_), blk, 0, stream>>>(Qh, Kh, trust, mask, stats, attn);
  k_rowstats<<<dim3((B_ * H_ * S_) / 8), blk, 0, stream>>>(attn, rowmax, rowinv);
  k_av<<<dim3(8, B_ * H_), blk, 0, stream>>>(Vt, rowmax, rowinv, attn, attOut);
  k_gate<<<dim3(8, 32), blk, 0, stream>>>(attOut, Wgt, Wg + (size_t)D_ * D_, bg, trust, gated);
  k_out<<<dim3(8, 32), blk, 0, stream>>>(gated, Wot, bo, out);
}
